// Attention_64269890617875
// MI455X (gfx1250) — compile-verified, matrix operand order checked by reading, not run on hardware
//
#include <hip/hip_runtime.h>
#include <hip/hip_bf16.h>
#include <math.h>
#include <stdint.h>

typedef _Float16 half_t;
typedef __attribute__((ext_vector_type(16))) _Float16 v16h;
typedef __attribute__((ext_vector_type(8)))  _Float16 v8h;
typedef __attribute__((ext_vector_type(8)))  float    v8f;
typedef __attribute__((ext_vector_type(4)))  unsigned v4u;
typedef __attribute__((ext_vector_type(8)))  int      v8i_t;
typedef __attribute__((ext_vector_type(4)))  int      v4i_t;

#define S_IMG 2048
#define S_TXT 512
#define S_TOTAL 2560
#define DMODEL 1536
#define NHEAD 24
#define DHEAD 64

// TDM availability (this toolchain: 6-arg builtin, therock-10.0 style)
#if defined(__has_builtin)
#if __has_builtin(__builtin_amdgcn_tensor_load_to_lds) && \
    __has_builtin(__builtin_amdgcn_s_wait_tensorcnt)
#define CDNA5_HAS_TDM 1
#endif
#endif
#ifndef CDNA5_HAS_TDM
#define CDNA5_HAS_TDM 0
#endif

// ---------------------------------------------------------------------------
// WMMA helper: D = A(16x32 f16) * B(32x16 f16) + C(16x16 f32)
// ---------------------------------------------------------------------------
__device__ __forceinline__ v8f wmma_f16(v16h a, v16h b, v8f c) {
  return __builtin_amdgcn_wmma_f32_16x16x32_f16(false, a, false, b,
                                                (short)0, c, false, false);
}

// LDS byte offset of a generic pointer to __shared__ (low 32 bits of the
// flat address are the LDS offset in the LDS aperture).
__device__ __forceinline__ unsigned lds_off(const void* p) {
  return (unsigned)(uintptr_t)p;
}

// Async global -> LDS copy, 16 bytes per lane. Tracked by ASYNCcnt.
__device__ __forceinline__ void async_ld_b128(unsigned lds_byte_off, const void* gaddr) {
  asm volatile("global_load_async_to_lds_b128 %0, %1, off"
               :: "v"(lds_byte_off), "v"(gaddr) : "memory");
}
__device__ __forceinline__ void wait_async0() {
  asm volatile("s_wait_asynccnt 0x0" ::: "memory");
}

#if CDNA5_HAS_TDM
// TDM 2D tile load: rows x cols (elements of 2 bytes) from a row-major matrix
// with row length row_stride_elems, into LDS at lds_byte. LDS padding: 4 DWORDs
// inserted after every 16 DWORDs => effective LDS row stride 40 halfs, matching
// the padded fragment layout. One instruction, tracked by TENSORcnt.
__device__ __forceinline__ void tdm_load_tile_2d(unsigned lds_byte, const void* gaddr,
                                                 int row_stride_elems, int rows, int cols) {
  const uint64_t ga = (uint64_t)(uintptr_t)gaddr;
  const unsigned td0 = (unsigned)row_stride_elems;   // tensor_dim0 (elems)
  const unsigned td1 = 0x100000u;                    // tensor_dim1 (large, OOB only)
  v4u g0 = {
      1u,                                            // count=1 (valid user D#)
      lds_byte,                                      // lds_addr
      (unsigned)(ga & 0xFFFFFFFFu),                  // global_addr[31:0]
      (unsigned)((ga >> 32) & 0x1FFFFFFu) | (2u << 30)  // addr[56:32], type=2
  };
  v8i_t g1 = {
      (int)((1u << 16) | (1u << 20) | (3u << 22) | (3u << 25)), // 2B, pad 16dw->+4dw
      (int)((td0 & 0xFFFFu) << 16),                             // td0[15:0]
      (int)(((td0 >> 16) & 0xFFFFu) | ((td1 & 0xFFFFu) << 16)), // td0 hi | td1 lo
      (int)(((td1 >> 16) & 0xFFFFu) | ((unsigned)cols << 16)),  // td1 hi | tile_dim0
      (int)(rows & 0xFFFF),                                     // tile_dim1 (dim2=0)
      (int)td0,                                                 // dim0_stride[31:0]
      0, 0
  };
  v4i_t z4 = {};
  v8i_t z8 = {};
  __builtin_amdgcn_tensor_load_to_lds(g0, g1, z4, z4, z8, 0);
}
#endif

// Fragment in the CDNA5 16-bit A/B register layout from a row-major
// 16 x (>=32) tile (index r = lane%16 selects row (A) / col (B), contraction
// runs contiguously in memory). ld in halfs.
template <typename P>
__device__ __forceinline__ v16h load_frag16x32(P base, int ld) {
  const int lane = threadIdx.x & 31;
  const int r = lane & 15;
  const int koff = (lane < 16) ? 0 : 8;
  P p = base + r * ld + koff;
  v16h f;
#pragma unroll
  for (int i = 0; i < 8; ++i) { f[i] = p[i]; f[i + 8] = p[i + 16]; }
  return f;
}

// B fragment when the tile is stored [K][N] row-major (contraction is the
// slow axis): col n = lane%16, contraction strided by ld.
template <typename P>
__device__ __forceinline__ v16h load_fragB_kmajor(P base, int ld) {
  const int lane = threadIdx.x & 31;
  const int n = lane & 15;
  const int koff = (lane < 16) ? 0 : 8;
  P p = base + n + koff * ld;
  v16h f;
#pragma unroll
  for (int i = 0; i < 8; ++i) { f[i] = p[i * ld]; f[i + 8] = p[(i + 16) * ld]; }
  return f;
}

// ---------------------------------------------------------------------------
// fp32 -> fp16 conversion
// ---------------------------------------------------------------------------
__global__ void cvt_f32_f16(const float* __restrict__ x, half_t* __restrict__ y, int n) {
  int i = blockIdx.x * blockDim.x + threadIdx.x;
  if (i < n) y[i] = (half_t)x[i];
}

// ---------------------------------------------------------------------------
// C[M,N] = A[M,K] @ W[N,K]^T + bias[N]
// 128x128 block tile, 8 waves (256 thr), each wave owns 64x32 (4x2 WMMA
// accumulators -> 8 WMMAs / k-step). K-step 32. Double-buffered LDS filled
// by the Tensor Data Mover (wave 0 issues one descriptor per operand tile;
// LDS pad feature produces the 40-half padded row stride). Async fallback.
// ---------------------------------------------------------------------------
__global__ void __launch_bounds__(256) gemm_xwT_bias(
    const half_t* __restrict__ A, const half_t* __restrict__ W,
    const float* __restrict__ bias, float* __restrict__ C,
    int M, int N, int K) {
  __shared__ __align__(16) half_t sA[2][128 * 40];
  __shared__ __align__(16) half_t sB[2][128 * 40];
  const int tid = threadIdx.x;
  const int wave = tid >> 5, lane = tid & 31;
  const int m0 = blockIdx.x * 128, n0 = blockIdx.y * 128;
  const int wr = (wave >> 2) * 64;   // 2 wave-rows x 64
  const int wc = (wave & 3) * 32;    // 4 wave-cols x 32
  v8f acc[4][2] = {};

  auto issue = [&](int buf, int k0) {
#if CDNA5_HAS_TDM
    if (wave == 0) {
      tdm_load_tile_2d(lds_off(&sA[buf][0]), &A[(size_t)m0 * K + k0], K, 128, 32);
      tdm_load_tile_2d(lds_off(&sB[buf][0]), &W[(size_t)n0 * K + k0], K, 128, 32);
    }
#else
#pragma unroll
    for (int i = tid; i < 512; i += 256) {   // 128 rows x 4 chunks, A and B
      int r = i >> 2, c = (i & 3) * 8;
      async_ld_b128(lds_off(&sA[buf][r * 40 + c]),
                    &A[(size_t)(m0 + r) * K + k0 + c]);
      async_ld_b128(lds_off(&sB[buf][r * 40 + c]),
                    &W[(size_t)(n0 + r) * K + k0 + c]);
    }
#endif
  };
  auto wait_stage = [&]() {
#if CDNA5_HAS_TDM
    if (wave == 0) __builtin_amdgcn_s_wait_tensorcnt((short)0);
#else
    wait_async0();
#endif
  };

  issue(0, 0);
  int cur = 0;
  for (int k0 = 0; k0 < K; k0 += 32) {
    wait_stage();           // staging into sX[cur] complete (issuing wave)
    __syncthreads();        // visible to all waves
    if (k0 + 32 < K) issue(cur ^ 1, k0 + 32);
    v16h a[4], b[2];
#pragma unroll
    for (int mt = 0; mt < 4; ++mt)
      a[mt] = load_frag16x32(&sA[cur][(wr + mt * 16) * 40], 40);
#pragma unroll
    for (int nt = 0; nt < 2; ++nt)
      b[nt] = load_frag16x32(&sB[cur][(wc + nt * 16) * 40], 40);
#pragma unroll
    for (int mt = 0; mt < 4; ++mt)
#pragma unroll
      for (int nt = 0; nt < 2; ++nt)
        acc[mt][nt] = wmma_f16(a[mt], b[nt], acc[mt][nt]);
    __syncthreads();        // done reading sX[cur]; safe to refill later
    cur ^= 1;
  }

  // C/D layout: VGPR i holds row i (lanes 0..15) / row i+8 (lanes 16..31)
  const int rbase = (lane < 16) ? 0 : 8;
  const int col = lane & 15;
#pragma unroll
  for (int mt = 0; mt < 4; ++mt)
#pragma unroll
    for (int nt = 0; nt < 2; ++nt)
#pragma unroll
      for (int i = 0; i < 8; ++i) {
        int row = m0 + wr + mt * 16 + rbase + i;
        int cc = n0 + wc + nt * 16 + col;
        C[(size_t)row * N + cc] = acc[mt][nt][i] + bias[cc];
      }
}

// ---------------------------------------------------------------------------
// Per-(seq,head) epilogue: RMS norm (+gain), RoPE, DH^-0.5 folded into Q.
// Repack to head-major f16: Q/K/V [H][S_TOT][64]. One wave per (s,h);
// lane owns the rotary pair (2*lane, 2*lane+1).
// ---------------------------------------------------------------------------
__global__ void __launch_bounds__(32) qkv_post(
    const float* __restrict__ qpi, const float* __restrict__ kpi, const float* __restrict__ vpi,
    const float* __restrict__ qpt, const float* __restrict__ kpt, const float* __restrict__ vpt,
    const float* __restrict__ ropec, const float* __restrict__ ropes,
    const float* __restrict__ gq, const float* __restrict__ gk,
    const float* __restrict__ gaq, const float* __restrict__ gak,
    half_t* __restrict__ Q, half_t* __restrict__ K, half_t* __restrict__ V) {
  const int s = blockIdx.x, h = blockIdx.y, lane = threadIdx.x;
  const bool txt = (s < S_TXT);
  const int r = txt ? s : s - S_TXT;
  const size_t ro = (size_t)r * DMODEL + h * DHEAD;
  const float* qrow = (txt ? qpt : qpi) + ro;
  const float* krow = (txt ? kpt : kpi) + ro;
  const float* vrow = (txt ? vpt : vpi) + ro;
  const float* gqv = txt ? gaq : gq;
  const float* gkv = txt ? gak : gk;
  const int e0 = 2 * lane, e1 = e0 + 1;
  float q0 = qrow[e0], q1 = qrow[e1];
  float k0 = krow[e0], k1 = krow[e1];
  float v0 = vrow[e0], v1 = vrow[e1];
  float qs = q0 * q0 + q1 * q1;
  float ks = k0 * k0 + k1 * k1;
#pragma unroll
  for (int mk = 16; mk >= 1; mk >>= 1) {
    qs += __shfl_xor(qs, mk, 32);
    ks += __shfl_xor(ks, mk, 32);
  }
  const float qr = rsqrtf(qs * (1.0f / DHEAD) + 1e-6f);
  const float kr = rsqrtf(ks * (1.0f / DHEAD) + 1e-6f);
  q0 *= qr * gqv[e0]; q1 *= qr * gqv[e1];
  k0 *= kr * gkv[e0]; k1 *= kr * gkv[e1];
  const float c  = ropec[(size_t)s * DHEAD + e0];  // repeated pairs: [2i]==[2i+1]
  const float sn = ropes[(size_t)s * DHEAD + e0];
  const float Q0 = (q0 * c - q1 * sn) * 0.125f;    // DH^-0.5 = 1/8 folded into Q
  const float Q1 = (q1 * c + q0 * sn) * 0.125f;
  const float K0 = k0 * c - k1 * sn;
  const float K1 = k1 * c + k0 * sn;
  const size_t o = ((size_t)h * S_TOTAL + s) * DHEAD;
  Q[o + e0] = (half_t)Q0; Q[o + e1] = (half_t)Q1;
  K[o + e0] = (half_t)K0; K[o + e1] = (half_t)K1;
  V[o + e0] = (half_t)v0; V[o + e1] = (half_t)v1;
}

// ---------------------------------------------------------------------------
// Flash attention, one block = (64 q-rows, 1 head), 4 waves x 16 q-rows.
// K/V tiles staged by async global->LDS copies (row-major [key][dh]).
// Per 32-key tile: S = Q Kt (4 WMMAs), streaming softmax with 16-lane
// shuffle row reductions, P transposed via per-wave LDS patch, O += P V
// (4 WMMAs; V consumed through a k-major strided B fragment).
// Output: fp32 [S_TOT][DMODEL] row-major (head h -> cols h*64..h*64+63).
// ---------------------------------------------------------------------------
__global__ void __launch_bounds__(128) flash_attn(
    const half_t* __restrict__ Q, const half_t* __restrict__ K,
    const half_t* __restrict__ V, float* __restrict__ O) {
  __shared__ __align__(16) half_t sK[32 * 72];    // [key][dh], pad 8
  __shared__ __align__(16) half_t sV[32 * 72];    // [key][dh], pad 8
  __shared__ __align__(16) half_t sP[4][16 * 40]; // per-wave P patch 16x32
  const int tid = threadIdx.x, wave = tid >> 5, lane = tid & 31;
  const int h = blockIdx.y;
  const int q0r = blockIdx.x * 64 + wave * 16;
  const half_t* Qh = Q + (size_t)h * S_TOTAL * DHEAD;
  const half_t* Kh = K + (size_t)h * S_TOTAL * DHEAD;
  const half_t* Vh = V + (size_t)h * S_TOTAL * DHEAD;

  const v16h qf0 = load_frag16x32(Qh + (size_t)q0r * DHEAD, DHEAD);       // dh 0..31
  const v16h qf1 = load_frag16x32(Qh + (size_t)q0r * DHEAD + 32, DHEAD);  // dh 32..63
  v8f acc[4] = {};
  float mrow[8], lrow[8];
#pragma unroll
  for (int i = 0; i < 8; ++i) { mrow[i] = -1e30f; lrow[i] = 0.0f; }

  const int rbase = (lane < 16) ? 0 : 8;
  const int cidx = lane & 15;

  for (int kb = 0; kb < S_TOTAL; kb += 32) {
    // async stage: K and V tiles, 32 rows x 64 halfs each
#pragma unroll
    for (int i = tid; i < 256; i += 128) {
      int r = i >> 3, c = (i & 7) * 8;
      async_ld_b128(lds_off(&sK[r * 72 + c]), &Kh[(size_t)(kb + r) * DHEAD + c]);
      async_ld_b128(lds_off(&sV[r * 72 + c]), &Vh[(size_t)(kb + r) * DHEAD + c]);
    }
    wait_async0();
    __syncthreads();

    // scores: s0 = keys kb..kb+15, s1 = keys kb+16..kb+31
    v8f s0 = {}, s1 = {};
    {
      v16h b;
      b = load_frag16x32(&sK[0], 72);            s0 = wmma_f16(qf0, b, s0);
      b = load_frag16x32(&sK[32], 72);           s0 = wmma_f16(qf1, b, s0);
      b = load_frag16x32(&sK[16 * 72], 72);      s1 = wmma_f16(qf0, b, s1);
      b = load_frag16x32(&sK[16 * 72 + 32], 72); s1 = wmma_f16(qf1, b, s1);
    }

    // streaming softmax per row (row = rbase + i, 16 lanes per row)
    float scale[8];
#pragma unroll
    for (int i = 0; i < 8; ++i) {
      float mi = fmaxf(s0[i], s1[i]);
#pragma unroll
      for (int mk = 8; mk >= 1; mk >>= 1) mi = fmaxf(mi, __shfl_xor(mi, mk, 32));
      float mn = fmaxf(mrow[i], mi);
      float sc = __expf(mrow[i] - mn);
      float p0 = __expf(s0[i] - mn);
      float p1 = __expf(s1[i] - mn);
      float ps = p0 + p1;
#pragma unroll
      for (int mk = 8; mk >= 1; mk >>= 1) ps += __shfl_xor(ps, mk, 32);
      mrow[i] = mn;
      lrow[i] = lrow[i] * sc + ps;
      scale[i] = sc;
      sP[wave][(rbase + i) * 40 + cidx] = (half_t)p0;
      sP[wave][(rbase + i) * 40 + 16 + cidx] = (half_t)p1;
    }
#pragma unroll
    for (int t = 0; t < 4; ++t)
#pragma unroll
      for (int i = 0; i < 8; ++i) acc[t][i] *= scale[i];

    asm volatile("s_wait_dscnt 0x0" ::: "memory");  // P patch visible to own wave
    v16h pa = load_frag16x32(&sP[wave][0], 40);     // A fragment 16x32 over keys
#pragma unroll
    for (int t = 0; t < 4; ++t) {
      v16h vb = load_fragB_kmajor(&sV[t * 16], 72); // B: col=dh (t*16+n), K=key strided
      acc[t] = wmma_f16(pa, vb, acc[t]);
    }
    __syncthreads();
  }

#pragma unroll
  for (int t = 0; t < 4; ++t)
#pragma unroll
    for (int i = 0; i < 8; ++i) {
      int srow = q0r + rbase + i;
      int dh = t * 16 + cidx;
      O[(size_t)srow * DMODEL + h * DHEAD + dh] = acc[t][i] / lrow[i];
    }
}

// ---------------------------------------------------------------------------
// Host orchestration
// ---------------------------------------------------------------------------
extern "C" void kernel_launch(void* const* d_in, const int* in_sizes, int n_in,
                              void* d_out, int out_size, void* d_ws, size_t ws_size,
                              hipStream_t stream) {
  const float* hid   = (const float*)d_in[0];
  const float* ehid  = (const float*)d_in[1];
  const float* ropec = (const float*)d_in[2];
  const float* ropes = (const float*)d_in[3];
  const float* Wq  = (const float*)d_in[4];   const float* bq  = (const float*)d_in[5];
  const float* Wk  = (const float*)d_in[6];   const float* bk  = (const float*)d_in[7];
  const float* Wv  = (const float*)d_in[8];   const float* bv  = (const float*)d_in[9];
  const float* Waq = (const float*)d_in[10];  const float* baq = (const float*)d_in[11];
  const float* Wak = (const float*)d_in[12];  const float* bak = (const float*)d_in[13];
  const float* Wav = (const float*)d_in[14];  const float* bav = (const float*)d_in[15];
  const float* Wo  = (const float*)d_in[16];  const float* bo  = (const float*)d_in[17];
  const float* Wao = (const float*)d_in[18];  const float* bao = (const float*)d_in[19];
  const float* gq  = (const float*)d_in[20];  const float* gk  = (const float*)d_in[21];
  const float* gaq = (const float*)d_in[22];  const float* gak = (const float*)d_in[23];

  float* out_img = (float*)d_out;                             // [2048,1536]
  float* out_txt = (float*)d_out + (size_t)S_IMG * DMODEL;    // [512,1536]

  // bump allocator over workspace
  uintptr_t cur = (uintptr_t)d_ws;
  auto alloc = [&](size_t bytes) -> void* {
    cur = (cur + 255) & ~(uintptr_t)255;
    void* p = (void*)cur;
    cur += bytes;
    return p;
  };

  const size_t nW = (size_t)DMODEL * DMODEL;
  half_t* Xi = (half_t*)alloc((size_t)S_IMG * DMODEL * 2);
  half_t* Xt = (half_t*)alloc((size_t)S_TXT * DMODEL * 2);
  half_t* hWq  = (half_t*)alloc(nW * 2);
  half_t* hWk  = (half_t*)alloc(nW * 2);
  half_t* hWv  = (half_t*)alloc(nW * 2);
  half_t* hWaq = (half_t*)alloc(nW * 2);
  half_t* hWak = (half_t*)alloc(nW * 2);
  half_t* hWav = (half_t*)alloc(nW * 2);
  half_t* hWo  = (half_t*)alloc(nW * 2);
  half_t* hWao = (half_t*)alloc(nW * 2);
  const size_t nQKV = (size_t)NHEAD * S_TOTAL * DHEAD;
  half_t* Qh = (half_t*)alloc(nQKV * 2);
  half_t* Kh = (half_t*)alloc(nQKV * 2);
  half_t* Vh = (half_t*)alloc(nQKV * 2);

  // shared region: 6 fp32 projection buffers, later reused for attention out
  char* reg = (char*)alloc((size_t)(3 * S_IMG + 3 * S_TXT) * DMODEL * 4);
  float* qpi = (float*)reg;
  float* kpi = qpi + (size_t)S_IMG * DMODEL;
  float* vpi = kpi + (size_t)S_IMG * DMODEL;
  float* qpt = vpi + (size_t)S_IMG * DMODEL;
  float* kpt = qpt + (size_t)S_TXT * DMODEL;
  float* vpt = kpt + (size_t)S_TXT * DMODEL;
  // aliases (valid after qkv_post consumed the projections)
  float*  attnO  = (float*)reg;                                   // [2560,1536] f32
  half_t* attnOh = (half_t*)(reg + (size_t)S_TOTAL * DMODEL * 4); // [2560,1536] f16

  auto cvt = [&](const float* src, half_t* dst, size_t n) {
    cvt_f32_f16<<<dim3((unsigned)((n + 255) / 256)), dim3(256), 0, stream>>>(src, dst, (int)n);
  };

  // 1) fp16 conversions
  cvt(hid, Xi, (size_t)S_IMG * DMODEL);
  cvt(ehid, Xt, (size_t)S_TXT * DMODEL);
  cvt(Wq, hWq, nW);   cvt(Wk, hWk, nW);   cvt(Wv, hWv, nW);
  cvt(Waq, hWaq, nW); cvt(Wak, hWak, nW); cvt(Wav, hWav, nW);
  cvt(Wo, hWo, nW);   cvt(Wao, hWao, nW);

  // 2) QKV projections (WMMA, 128x128 tiles, TDM-staged double buffered)
  dim3 blk(256);
  dim3 gImg(S_IMG / 128, DMODEL / 128), gTxt(S_TXT / 128, DMODEL / 128);
  gemm_xwT_bias<<<gImg, blk, 0, stream>>>(Xi, hWq, bq, qpi, S_IMG, DMODEL, DMODEL);
  gemm_xwT_bias<<<gImg, blk, 0, stream>>>(Xi, hWk, bk, kpi, S_IMG, DMODEL, DMODEL);
  gemm_xwT_bias<<<gImg, blk, 0, stream>>>(Xi, hWv, bv, vpi, S_IMG, DMODEL, DMODEL);
  gemm_xwT_bias<<<gTxt, blk, 0, stream>>>(Xt, hWaq, baq, qpt, S_TXT, DMODEL, DMODEL);
  gemm_xwT_bias<<<gTxt, blk, 0, stream>>>(Xt, hWak, bak, kpt, S_TXT, DMODEL, DMODEL);
  gemm_xwT_bias<<<gTxt, blk, 0, stream>>>(Xt, hWav, bav, vpt, S_TXT, DMODEL, DMODEL);

  // 3) RMS + RoPE + repack
  qkv_post<<<dim3(S_TOTAL, NHEAD), dim3(32), 0, stream>>>(
      qpi, kpi, vpi, qpt, kpt, vpt, ropec, ropes, gq, gk, gaq, gak, Qh, Kh, Vh);

  // 4) attention (projection buffers dead -> region reused for attnO)
  flash_attn<<<dim3(S_TOTAL / 64, NHEAD), dim3(128), 0, stream>>>(Qh, Kh, Vh, attnO);

  // 5) fp16 copy of attention output
  cvt(attnO, attnOh, (size_t)S_TOTAL * DMODEL);

  // 6) output projections directly into d_out (text rows first in attnO)
  gemm_xwT_bias<<<gImg, blk, 0, stream>>>(attnOh + (size_t)S_TXT * DMODEL, hWo, bo,
                                          out_img, S_IMG, DMODEL, DMODEL);
  gemm_xwT_bias<<<gTxt, blk, 0, stream>>>(attnOh, hWao, bao,
                                          out_txt, S_TXT, DMODEL, DMODEL);
}
